// DPCNN_22960895164478
// MI455X (gfx1250) — compile-verified, matrix operand order checked by reading, not run on hardware
//
#include <hip/hip_runtime.h>
#include <cstdint>

typedef __attribute__((ext_vector_type(16))) _Float16 v16h;
typedef __attribute__((ext_vector_type(8)))  float    v8f;

#define DEVFN static __device__ __forceinline__

#if defined(__has_builtin)
#if __has_builtin(__builtin_amdgcn_global_load_async_to_lds_b32)
#define HAVE_ASYNC_LDS 1
#endif
#endif
#ifndef HAVE_ASYNC_LDS
#define HAVE_ASYNC_LDS 0
#endif

// 120 independent 100x500 single-channel images (B=8 x CHUNKS=15)
static constexpr int S = 120;
static constexpr int L1_CIN=1,  L1_H=100, L1_W=500, L1_COUT=16, L1_KP=32,   L1_PH=50, L1_PW=250;
static constexpr int L2_CIN=16, L2_H=50,  L2_W=250, L2_COUT=32, L2_KP=416,  L2_PH=25, L2_PW=125;
static constexpr int L3_CIN=32, L3_H=25,  L3_W=125, L3_COUT=64, L3_KP=800,  L3_PH=12, L3_PW=62;
static constexpr int L4_CIN=64, L4_H=12,  L4_W=62,  L4_COUT=32, L4_KP=1600, L4_PH=6,  L4_PW=31;
static constexpr int FC_K = 32*6*31; // 5952

// ---------------------------------------------------------------------------
// async global -> LDS dword copy (ASYNCcnt path) with synchronous fallback
// ---------------------------------------------------------------------------
DEVFN void copy_b32_g2l(_Float16* ldst, const _Float16* gsrc) {
#if HAVE_ASYNC_LDS
  __builtin_amdgcn_global_load_async_to_lds_b32(
      (__attribute__((address_space(1))) int*)gsrc,
      (__attribute__((address_space(3))) int*)ldst, 0, 0);
#else
  *(unsigned int*)ldst = *(const unsigned int*)gsrc;
#endif
}

DEVFN void wait_async0() {
#if HAVE_ASYNC_LDS
#if __has_builtin(__builtin_amdgcn_s_wait_asynccnt)
  __builtin_amdgcn_s_wait_asynccnt(0);
#else
  asm volatile("s_wait_asynccnt 0x0" ::: "memory");
#endif
#endif
}

// ---------------------------------------------------------------------------
// Fragment loader for 16-bit A/B of V_WMMA_F32_16X16X32_F16 (ISA 7.12.2):
// per lane, 16 halves at K offsets {lo..lo+7, lo+16..lo+23}, lo = hiHalf?8:0,
// packed pairwise into 8 VGPRs. p must be 4-byte aligned, contiguous in K.
// ---------------------------------------------------------------------------
DEVFN v16h load_frag_k32(const _Float16* p, int hiHalf) {
  const unsigned int* p32 = (const unsigned int*)p;
  const int lo2 = hiHalf * 4; // dword index of K=lo
  v16h f;
#pragma unroll
  for (int j = 0; j < 4; ++j) {
    union { unsigned int u; _Float16 h[2]; } c0, c1;
    c0.u = p32[lo2 + j];
    c1.u = p32[lo2 + 8 + j];
    f[2*j]       = c0.h[0];
    f[2*j + 1]   = c0.h[1];
    f[8 + 2*j]   = c1.h[0];
    f[8 + 2*j+1] = c1.h[1];
  }
  return f;
}

// ---------------------------------------------------------------------------
// prep: chunk-split input, f32 -> f16.  xh[s][h][w], s = b*15 + chunk
// ---------------------------------------------------------------------------
__global__ void prep_x_kernel(const float* __restrict__ x, _Float16* __restrict__ xh, int total) {
  int o = blockIdx.x * blockDim.x + threadIdx.x;
  if (o >= total) return;
  int w = o % 500;
  int t = o / 500;
  int h = t % 100;
  int s = t / 100;
  int b = s / 15, c = s - b*15;
  xh[o] = (_Float16)x[((size_t)b*100 + h)*7500 + c*500 + w];
}

// ---------------------------------------------------------------------------
// prep: plain f32 -> f16 conversion (used for wfc)
// ---------------------------------------------------------------------------
__global__ void prep_h_kernel(const float* __restrict__ src, _Float16* __restrict__ dst, int total) {
  int i = blockIdx.x * blockDim.x + threadIdx.x;
  if (i < total) dst[i] = (_Float16)src[i];
}

// ---------------------------------------------------------------------------
// prep: fold eval-BN into conv weights, convert f16, pad K to Kpad with zeros
// ---------------------------------------------------------------------------
__global__ void prep_w_kernel(const float* __restrict__ w, const float* __restrict__ b,
                              const float* __restrict__ g, const float* __restrict__ beta,
                              const float* __restrict__ m, const float* __restrict__ v,
                              _Float16* __restrict__ wp, float* __restrict__ bp,
                              int Cin, int Cout, int Kpad) {
  int idx = blockIdx.x * blockDim.x + threadIdx.x;
  if (idx >= Cout * Kpad) return;
  int co = idx / Kpad, k = idx - co*Kpad;
  float scale = g[co] * rsqrtf(v[co] + 1e-5f);
  wp[idx] = (k < Cin*25) ? (_Float16)(w[co*Cin*25 + k] * scale) : (_Float16)0.f;
  if (k == 0) bp[co] = (b[co] - m[co]) * scale + beta[co];
}

// ---------------------------------------------------------------------------
// Implicit-GEMM conv(5x5,pad2) + folded BN + ReLU + fused 2x2 maxpool.
// One wave: 16 Cout x 16 pooled outputs of one pooled row. Receptive field
// (CIN x 7 x 37 halves) staged once into LDS (zero-padded, async where
// aligned). K loop: one A fragment, 4 WMMAs (one per pool offset), LDS-only
// B gather with compile-time strides. No barriers inside the K loop.
// ---------------------------------------------------------------------------
template <int CIN, int HIN, int WIN, int COUT, int KPAD, int PH, int PW>
__global__ void __launch_bounds__(32)
conv_wmma_t(const _Float16* __restrict__ in, const _Float16* __restrict__ wp,
            const float* __restrict__ bp, _Float16* __restrict__ out)
{
  constexpr int CINK = CIN * 25;
  constexpr bool FULLK = (KPAD == CINK);
  constexpr int PWT = (PW + 15) / 16;
  constexpr int PR  = 7;                 // patch rows
  constexpr int PWW = 40;                // patch width (37 used, padded)
  constexpr int HW  = HIN * WIN;
  constexpr int PRW2 = PR * (PWW/2);     // dword pairs per channel

  const int lane = threadIdx.x;
  const int s = blockIdx.z;
  const int ph = blockIdx.y;
  const int pwBase = (blockIdx.x % PWT) * 16;
  const int coutBase = (blockIdx.x / PWT) * 16;

  __shared__ __align__(16) _Float16 patch[CIN * PR * PWW];

  const _Float16* inS = in + (size_t)s * CIN * HW;
  const int ih0 = 2*ph - 2;
  const int iw0 = 2*pwBase - 2;

  // ---- stage zero-padded input patch into LDS ----
  unsigned int* patch32 = (unsigned int*)patch;
  for (int e = lane; e < CIN * PRW2; e += 32) {
    int ci = e / PRW2;
    int rem = e - ci * PRW2;
    int r = rem / (PWW/2);
    int jp = rem - r * (PWW/2);
    int ih = ih0 + r;
    int iw = iw0 + 2*jp;
    const _Float16* src = inS + (size_t)ci*HW + (size_t)ih*WIN + iw;
    bool rowOK = (unsigned)ih < (unsigned)HIN;
    if (rowOK && iw >= 0 && (iw + 1) < WIN && ((((size_t)src) & 3) == 0)) {
      copy_b32_g2l(&patch[e*2], src);                 // async global->LDS b32
    } else {
      union { unsigned int u; _Float16 h[2]; } c; c.u = 0;
      if (rowOK) {
        if ((unsigned)iw     < (unsigned)WIN) c.h[0] = src[0];
        if ((unsigned)(iw+1) < (unsigned)WIN) c.h[1] = src[1];
      }
      patch32[e] = c.u;
    }
  }
  wait_async0();
  __syncthreads();

  const int nlane = lane & 15;
  const int hiHalf = (lane >> 4) & 1;
  const int lo = hiHalf ? 8 : 0;

  v8f zero8 = {};
  v8f accs[4] = {zero8, zero8, zero8, zero8};

  for (int k0 = 0; k0 < KPAD; k0 += 32) {
    const _Float16* aRow = wp + (size_t)(coutBase + nlane) * KPAD + k0;
    __builtin_prefetch(aRow + 32, 0, 1);              // global_prefetch_b8
    v16h a = load_frag_k32(aRow, hiHalf);

    // decode 16 K indices once (constant divisors); shared across 4 offsets
    int off[16];
    bool valid[16];
#pragma unroll
    for (int t = 0; t < 16; ++t) {
      int Koff = (t < 8) ? (lo + t) : (lo + 8 + t);   // lo..lo+7, lo+16..lo+23
      int k = k0 + Koff;
      int ci = k / 25, r5 = k - ci*25, kh = r5 / 5, kw = r5 - kh*5;
      off[t] = ci * (PR*PWW) + kh * PWW + kw + 2*nlane;
      valid[t] = FULLK || (k < CINK);
    }
#pragma unroll
    for (int p = 0; p < 4; ++p) {
      const int dy = p >> 1, dx = p & 1;
      v16h b;
#pragma unroll
      for (int t = 0; t < 16; ++t) {
        _Float16 v = (_Float16)0.f;
        if (valid[t]) v = patch[off[t] + dy*PWW + dx];
        b[t] = v;
      }
      accs[p] = __builtin_amdgcn_wmma_f32_16x16x32_f16(false, a, false, b,
                                                       (short)0, accs[p], false, false);
    }
  }

  // max over pool window (commutes with monotone bias+ReLU), then store f16
  const int pw = pwBase + nlane;
  if (pw < PW) {
#pragma unroll
    for (int r = 0; r < 8; ++r) {
      int co = coutBase + r + hiHalf*8;
      float m01 = fmaxf(accs[0][r], accs[1][r]);
      float m23 = fmaxf(accs[2][r], accs[3][r]);
      float v = fmaxf(fmaxf(m01, m23) + bp[co], 0.f);
      out[(size_t)s * COUT * (PH*PW) + (size_t)co * (PH*PW) + ph*PW + pw] = (_Float16)v;
    }
  }
}

// ---------------------------------------------------------------------------
// FC: res[120x32] = feats[120x5952] @ wfc.T + bfc  (WMMA, M padded to 128)
// B[k][n] = wfcH[n][k]: a lane's B fragment (fixed N, K-contiguous pattern)
// is a straight dword-pattern load from one wfcH row — no LDS, no barriers.
// ---------------------------------------------------------------------------
__global__ void __launch_bounds__(32)
fc_wmma(const _Float16* __restrict__ feats, const _Float16* __restrict__ wfcH,
        const float* __restrict__ bfc, float* __restrict__ res)
{
  const int K = FC_K;
  const int lane = threadIdx.x;
  const int nbase = blockIdx.x * 16;
  const int mbase = blockIdx.y * 16;
  const int nlane = lane & 15;
  const int hiHalf = (lane >> 4) & 1;
  v8f acc = {};
  const int m_l = mbase + nlane;
  const _Float16* aBase = feats + (size_t)m_l * K;
  const _Float16* bBase = wfcH + (size_t)(nbase + nlane) * K;
  for (int k0 = 0; k0 < K; k0 += 32) {
    __builtin_prefetch(bBase + k0 + 64, 0, 1);
    v16h a = {};
    if (m_l < S) {
      __builtin_prefetch(aBase + k0 + 64, 0, 1);
      a = load_frag_k32(aBase + k0, hiHalf);
    }
    v16h b = load_frag_k32(bBase + k0, hiHalf);
    acc = __builtin_amdgcn_wmma_f32_16x16x32_f16(false, a, false, b,
                                                 (short)0, acc, false, false);
  }
#pragma unroll
  for (int r = 0; r < 8; ++r) {
    int mm = mbase + r + hiHalf*8;
    if (mm < S) res[mm*32 + nbase + nlane] = acc[r] + bfc[nbase + nlane];
  }
}

// ---------------------------------------------------------------------------
// DPCNN head: one wave per batch, all in LDS (f32, tiny)
// ---------------------------------------------------------------------------
DEVFN void head_conv(const float* in, int is, float* out, int os,
                     const float* w, const float* bias, int H, int lane) {
  for (int e = lane; e < 15*H; e += 32) {
    int c = e / H, h = e - c*H;
    float sacc = bias[c];
#pragma unroll
    for (int ci = 0; ci < 15; ++ci)
#pragma unroll
      for (int kh = 0; kh < 3; ++kh) {
        int hh = h - 1 + kh;
        if ((unsigned)hh < (unsigned)H)
          sacc += w[(c*15 + ci)*3 + kh] * fmaxf(in[ci*is + hh], 0.f);
      }
    out[c*os + h] = sacc;
  }
}

__global__ void __launch_bounds__(32)
head_dpcnn(const float* __restrict__ res, const float* __restrict__ wc3,
           const float* __restrict__ bc3, const float* __restrict__ wout,
           const float* __restrict__ bout, float* __restrict__ outp)
{
  const int b = blockIdx.x;
  const int lane = threadIdx.x;
  __shared__ float cur[15*34], tmp[15*34], px[15*17];
  __shared__ float wsh[15*15*3], bsh2[15];
  for (int e = lane; e < 675; e += 32) wsh[e] = wc3[e];
  for (int e = lane; e < 15;  e += 32) bsh2[e] = bc3[e];
  for (int e = lane; e < 15*32; e += 32)
    cur[(e >> 5)*34 + (e & 31)] = res[(b*15 + (e >> 5))*32 + (e & 31)];
  __syncthreads();
  head_conv(cur,34, tmp,34, wsh,bsh2, 32, lane); __syncthreads();
  head_conv(tmp,34, cur,34, wsh,bsh2, 32, lane); __syncthreads();
  int H = 32;
  while (H > 2) {
    int Hn = H >> 1;
    for (int e = lane; e < 15*Hn; e += 32) {     // pad-bottom(0) + maxpool(3, stride 2)
      int c = e / Hn, h = e - c*Hn;
      float a  = cur[c*34 + 2*h];
      float b2 = cur[c*34 + 2*h + 1];
      float c3 = (2*h + 2 < H) ? cur[c*34 + 2*h + 2] : 0.f;
      px[c*17 + h] = fmaxf(fmaxf(a, b2), c3);
    }
    __syncthreads();
    head_conv(px,17, tmp,34, wsh,bsh2, Hn, lane); __syncthreads();
    head_conv(tmp,34, cur,34, wsh,bsh2, Hn, lane); __syncthreads();
    for (int e = lane; e < 15*Hn; e += 32) {     // residual
      int c = e / Hn, h = e - c*Hn;
      cur[c*34 + h] += px[c*17 + h];
    }
    __syncthreads();
    H = Hn;
  }
  if (lane == 0) {
#pragma unroll
    for (int j = 0; j < 2; ++j) {
      float sacc = bout[j];
      for (int c = 0; c < 15; ++c)
        for (int h = 0; h < 2; ++h)
          sacc += wout[j*30 + c*2 + h] * cur[c*34 + h];
      outp[b*2 + j] = sacc;
    }
  }
}

// ---------------------------------------------------------------------------
extern "C" void kernel_launch(void* const* d_in, const int* in_sizes, int n_in,
                              void* d_out, int out_size, void* d_ws, size_t ws_size,
                              hipStream_t stream) {
  (void)in_sizes; (void)n_in; (void)out_size; (void)ws_size;
  const float* x = (const float*)d_in[0];
  const float *W[4], *Bb[4], *G[4], *Bt[4], *Mn[4], *Vr[4];
  for (int i = 0; i < 4; ++i) {
    W[i]  = (const float*)d_in[1 + 6*i];
    Bb[i] = (const float*)d_in[2 + 6*i];
    G[i]  = (const float*)d_in[3 + 6*i];
    Bt[i] = (const float*)d_in[4 + 6*i];
    Mn[i] = (const float*)d_in[5 + 6*i];
    Vr[i] = (const float*)d_in[6 + 6*i];
  }
  const float* wfc  = (const float*)d_in[25];
  const float* bfc  = (const float*)d_in[26];
  const float* wc3  = (const float*)d_in[27];
  const float* bc3  = (const float*)d_in[28];
  const float* wout = (const float*)d_in[29];
  const float* bout = (const float*)d_in[30];
  float* out = (float*)d_out;

  char* p = (char*)d_ws;
  auto take = [&](size_t bytes) { char* r = p; p += (bytes + 255) & ~(size_t)255; return r; };
  _Float16* xh   = (_Float16*)take((size_t)S * L1_H * L1_W * 2);
  _Float16* wp1  = (_Float16*)take((size_t)L1_COUT * L1_KP * 2);
  _Float16* wp2  = (_Float16*)take((size_t)L2_COUT * L2_KP * 2);
  _Float16* wp3  = (_Float16*)take((size_t)L3_COUT * L3_KP * 2);
  _Float16* wp4  = (_Float16*)take((size_t)L4_COUT * L4_KP * 2);
  float*    bp1  = (float*)take(L1_COUT * 4);
  float*    bp2  = (float*)take(L2_COUT * 4);
  float*    bp3  = (float*)take(L3_COUT * 4);
  float*    bp4  = (float*)take(L4_COUT * 4);
  _Float16* act1 = (_Float16*)take((size_t)S * L1_COUT * L1_PH * L1_PW * 2);
  _Float16* act2 = (_Float16*)take((size_t)S * L2_COUT * L2_PH * L2_PW * 2);
  _Float16* act3 = (_Float16*)take((size_t)S * L3_COUT * L3_PH * L3_PW * 2);
  _Float16* act4 = (_Float16*)take((size_t)S * L4_COUT * L4_PH * L4_PW * 2);
  _Float16* wfcH = (_Float16*)take((size_t)32 * FC_K * 2);
  float*    res  = (float*)take((size_t)S * 32 * 4);

  const int totalx = S * L1_H * L1_W;
  prep_x_kernel<<<(totalx + 255) / 256, 256, 0, stream>>>(x, xh, totalx);
  prep_h_kernel<<<(32*FC_K + 255) / 256, 256, 0, stream>>>(wfc, wfcH, 32*FC_K);
  prep_w_kernel<<<(L1_COUT*L1_KP + 255)/256, 256, 0, stream>>>(W[0],Bb[0],G[0],Bt[0],Mn[0],Vr[0], wp1,bp1, L1_CIN,L1_COUT,L1_KP);
  prep_w_kernel<<<(L2_COUT*L2_KP + 255)/256, 256, 0, stream>>>(W[1],Bb[1],G[1],Bt[1],Mn[1],Vr[1], wp2,bp2, L2_CIN,L2_COUT,L2_KP);
  prep_w_kernel<<<(L3_COUT*L3_KP + 255)/256, 256, 0, stream>>>(W[2],Bb[2],G[2],Bt[2],Mn[2],Vr[2], wp3,bp3, L3_CIN,L3_COUT,L3_KP);
  prep_w_kernel<<<(L4_COUT*L4_KP + 255)/256, 256, 0, stream>>>(W[3],Bb[3],G[3],Bt[3],Mn[3],Vr[3], wp4,bp4, L4_CIN,L4_COUT,L4_KP);

  conv_wmma_t<L1_CIN,L1_H,L1_W,L1_COUT,L1_KP,L1_PH,L1_PW>
      <<<dim3(((L1_PW+15)/16)*(L1_COUT/16), L1_PH, S), 32, 0, stream>>>(xh,   wp1, bp1, act1);
  conv_wmma_t<L2_CIN,L2_H,L2_W,L2_COUT,L2_KP,L2_PH,L2_PW>
      <<<dim3(((L2_PW+15)/16)*(L2_COUT/16), L2_PH, S), 32, 0, stream>>>(act1, wp2, bp2, act2);
  conv_wmma_t<L3_CIN,L3_H,L3_W,L3_COUT,L3_KP,L3_PH,L3_PW>
      <<<dim3(((L3_PW+15)/16)*(L3_COUT/16), L3_PH, S), 32, 0, stream>>>(act2, wp3, bp3, act3);
  conv_wmma_t<L4_CIN,L4_H,L4_W,L4_COUT,L4_KP,L4_PH,L4_PW>
      <<<dim3(((L4_PW+15)/16)*(L4_COUT/16), L4_PH, S), 32, 0, stream>>>(act3, wp4, bp4, act4);

  fc_wmma<<<dim3(2, (S + 15) / 16), 32, 0, stream>>>(act4, wfcH, bfc, res);
  head_dpcnn<<<8, 32, 0, stream>>>(res, wc3, bc3, wout, bout, out);
}
